// INT4Linear_14783277433034
// MI455X (gfx1250) — matrix operand order repaired and checked by reading, not compiled, not run on hardware
//
#include <hip/hip_runtime.h>
#include <hip/hip_bf16.h>

// ---- problem constants (from reference) ----
#define IN_F    4096
#define OUT_F   11008
#define QOUT    5504          // OUT_F/2 packed int4 bytes (one per int32)
#define GROUP   128
#define NGROUPS 32
#define BATCH   32

// ---- tiling ----
#define NBLK      128         // output columns per block (8 waves * 16)
#define PCBLK     64          // packed columns per block (dwords per tile row)
#define KCH       32          // K per WMMA step (bf16 16x16x32)
#define KTILE     64          // K rows staged per TDM tile (2 WMMA steps)
#define NTILES    (IN_F / KTILE)   // 64
#define ROWSTRIDE 65          // LDS dwords per row (64 + 1 TDM pad dword)

typedef __attribute__((ext_vector_type(16))) __bf16       v16bf;
typedef __attribute__((ext_vector_type(8)))  __bf16       v8bf;
typedef __attribute__((ext_vector_type(8)))  float        v8f;
typedef __attribute__((ext_vector_type(4)))  float        v4f;
typedef __attribute__((ext_vector_type(4)))  int          v4i;
typedef __attribute__((ext_vector_type(8)))  int          v8i;
typedef __attribute__((ext_vector_type(4)))  unsigned int v4u;

// ---- TDM availability / arity discrimination ----
#if defined(__has_builtin)
#  if __has_builtin(__builtin_amdgcn_tensor_load_to_lds) && \
      __has_builtin(__builtin_amdgcn_s_wait_tensorcnt)
#    define USE_TDM 1
#  endif
#endif
#ifndef USE_TDM
#  define USE_TDM 0
#endif
#if __has_include(<hip/amd_detail/amd_gfx1250_TDM.h>)
#  define TDM_6ARG 1            // therock-10.0 headers -> 6-arg builtin
#else
#  define TDM_6ARG 0            // ROCm 7.2 -> 5-arg builtin
#endif

#if USE_TDM
// One TDM 2D tile load: qweight[k0..k0+KTILE) x [gpc0..gpc0+64) dwords into LDS
// at lds_off, with 1 pad dword per 64-dword row (bank rotation for the gather).
static __device__ __forceinline__ void tdm_load_tile(const int* __restrict__ qw,
                                                     int k0, int gpc0,
                                                     unsigned lds_off) {
  unsigned long long ga =
      (unsigned long long)(uintptr_t)(qw + (size_t)k0 * QOUT + gpc0);
  v4u g0;
  g0[0] = 1u;                                        // count=1, user descriptor
  g0[1] = lds_off;                                   // LDS byte address
  g0[2] = (unsigned)(ga & 0xFFFFFFFFu);              // global addr lo
  g0[3] = (unsigned)((ga >> 32) & 0x1FFFFFFu)        // global addr hi [56:32]
          | (2u << 30);                              // type = 2 ("image")
  v8i g1;
  g1[0] = (2 << 16)        // data_size = 4 B
        | (1 << 20)        // pad_enable
        | (5 << 22)        // pad_interval: every 64 dwords
        | (0 << 25);       // pad_amount: 1 dword
  g1[1] = (int)((QOUT & 0xFFFF) << 16);                                // dim0 lo16
  g1[2] = (int)((QOUT >> 16) & 0xFFFF) | (int)((IN_F & 0xFFFF) << 16);  // dim0 hi | dim1 lo
  g1[3] = (int)((IN_F >> 16) & 0xFFFF) | (PCBLK << 16);                 // dim1 hi | tile_dim0
  g1[4] = KTILE;                                                        // tile_dim1
  g1[5] = QOUT;                                                         // dim0_stride lo32
  g1[6] = 0;
  g1[7] = 0;
  v4i z4 = {0, 0, 0, 0};
#if TDM_6ARG
  v8i z8 = {0, 0, 0, 0, 0, 0, 0, 0};
  __builtin_amdgcn_tensor_load_to_lds(g0, g1, z4, z4, z8, 0);
#else
  __builtin_amdgcn_tensor_load_to_lds(g0, g1, z4, z4, 0);
#endif
}
#endif // USE_TDM

static __device__ __forceinline__ v16bf load_a_frag(const v4i* __restrict__ p) {
  v4i x = p[0], y = p[1];
  v8i t;
  t[0] = x[0]; t[1] = x[1]; t[2] = x[2]; t[3] = x[3];
  t[4] = y[0]; t[5] = y[1]; t[6] = y[2]; t[7] = y[3];
  return __builtin_bit_cast(v16bf, t);
}

__global__ __launch_bounds__(256)
void int4linear_wmma(const float* __restrict__ inp,      // [32][4096]
                     const int*   __restrict__ qweight,  // [4096][5504]
                     const float* __restrict__ scales,   // [32][11008]
                     const int*   __restrict__ qzeros,   // [32][5504]
                     const float* __restrict__ bias,     // [11008]
                     float*       __restrict__ out) {    // [32][11008]
  __shared__ int   lds_q[2][KTILE * ROWSTRIDE];  // 33280 B: TDM weight tiles
  __shared__ v4i   lds_a[2 * 2 * 32 * 3];        //  6144 B: A bf16 frags (48B slots)
  __shared__ float lds_s[NGROUPS * BATCH];       //  4096 B: per-group row sums

  const int tid  = threadIdx.x;
  const int wave = tid >> 5;
  const int lane = tid & 31;
  const int c    = lane & 15;
  const int half = lane >> 4;

  const int n     = blockIdx.x * NBLK + wave * 16 + c;   // lane's out column
  const int pcl   = wave * 8 + (c >> 1);                 // packed col in tile
  const int shift = (c & 1) * 4;
  const int kbB   = half * 16;
  const int gpc0  = blockIdx.x * PCBLK;

  // ---- precompute S[g][b] = sum_{k in group g} inp[b][k]  (1024 values)
#pragma unroll
  for (int i = 0; i < 4; ++i) {
    const int idx = tid + i * 256;
    const int sg  = idx >> 5;
    const int sb  = idx & 31;
    const v4f* p  = (const v4f*)(inp + sb * IN_F + sg * GROUP);
    v4f a = {0.f, 0.f, 0.f, 0.f};
#pragma unroll 8
    for (int j = 0; j < GROUP / 4; ++j) a += p[j];
    lds_s[sg * BATCH + sb] = a[0] + a[1] + a[2] + a[3];
  }
  __syncthreads();

  v8f acc0 = {};
  v8f acc1 = {};

#if USE_TDM
  if (wave == 0) tdm_load_tile(qweight, 0, gpc0, (unsigned)(uintptr_t)&lds_q[0][0]);
#else
  const int frow = tid >> 2;          // fallback staging: 0..63
  const int fcol = (tid & 3) * 16;    // 0,16,32,48
#endif

  for (int g = 0; g < NGROUPS; ++g) {
    v8f P0 = {};                      // group partial: sum a*q, rows 0..15
    v8f P1 = {};                      // rows 16..31

#pragma unroll
    for (int ph = 0; ph < 2; ++ph) {
      const int t   = g * 2 + ph;
      const int buf = t & 1;
      const int kt0 = t * KTILE;

      __syncthreads();                // prior compute done with lds_q[buf], lds_a
#if USE_TDM
      if (wave == 0 && t + 1 < NTILES)
        tdm_load_tile(qweight, kt0 + KTILE, gpc0,
                      (unsigned)(uintptr_t)&lds_q[(t + 1) & 1][0]);
#else
      {
        const int* gp = qweight + (size_t)(kt0 + frow) * QOUT + gpc0 + fcol;
        if (t + 1 < NTILES)
          __builtin_prefetch(gp + KTILE * QOUT, 0, 0);
#pragma unroll
        for (int j = 0; j < 4; ++j) {
          v4i q = *(const v4i*)(gp + j * 4);
          *(v4i*)&lds_q[buf][frow * ROWSTRIDE + fcol + j * 4] = q;
        }
      }
#endif
      // ---- cooperative A staging: fp32 -> bf16 WMMA-A fragments (once per block)
      {
        const int part = tid & 1;          // 16B half of a 32B fragment
        const int slot = tid >> 1;         // 0..127: (kc,mtile,lane)
        const int al   = slot & 31;
        const int amt  = (slot >> 5) & 1;
        const int akc  = slot >> 6;
        const int arow = amt * 16 + (al & 15);
        const int akb  = (al >> 4) * 16;
        const float* src = inp + arow * IN_F + kt0 + akc * KCH + part * 8 + akb;
        v4f f0 = *(const v4f*)src;
        v4f f1 = *(const v4f*)(src + 4);
        v8bf hf;
#pragma unroll
        for (int i = 0; i < 4; ++i) {
          hf[i]     = (__bf16)f0[i];
          hf[4 + i] = (__bf16)f1[i];
        }
        lds_a[slot * 3 + part] = __builtin_bit_cast(v4i, hf);
      }
#if USE_TDM
      if (wave == 0) {
        if (t + 1 < NTILES) __builtin_amdgcn_s_wait_tensorcnt(1);  // tile t landed
        else                __builtin_amdgcn_s_wait_tensorcnt(0);
      }
#endif
      __syncthreads();                // weight tile + A fragments visible

#pragma unroll
      for (int kc = 0; kc < KTILE / KCH; ++kc) {
        v16bf A0 = load_a_frag(&lds_a[((kc * 2 + 0) * 32 + lane) * 3]);
        v16bf A1 = load_a_frag(&lds_a[((kc * 2 + 1) * 32 + lane) * 3]);

        // per-lane column gather: 16 strided dwords (bank-rotated rows)
        const int* lp = &lds_q[buf][(kc * KCH + kbB) * ROWSTRIDE + pcl];
        int bq[16];
#pragma unroll
        for (int e = 0; e < 16; ++e) bq[e] = lp[e * ROWSTRIDE];

        // raw nibble -> bf16 (exact); scale/zero folded at group flush
        v16bf B;
#pragma unroll
        for (int e = 0; e < 16; ++e)
          B[e] = (__bf16)(float)((((unsigned)bq[e]) >> shift) & 0xFu);

        P0 = __builtin_amdgcn_wmma_f32_16x16x32_bf16(false, A0, false, B,
                                                     (short)0, P0, false, false);
        P1 = __builtin_amdgcn_wmma_f32_16x16x32_bf16(false, A1, false, B,
                                                     (short)0, P1, false, false);
      }
    }

    // ---- group flush: acc += s*P + (-s*z)*S
    {
      const float s   = scales[g * OUT_F + n];
      const int   zq  = qzeros[g * QOUT + (n >> 1)];
      const float nzs = -(float)((((unsigned)zq) >> shift) & 0xFu) * s;
      const v4f* sp0 = (const v4f*)&lds_s[g * BATCH + half * 8];
      const v4f* sp1 = (const v4f*)&lds_s[g * BATCH + 16 + half * 8];
      v4f S0a = sp0[0], S0b = sp0[1];
      v4f S1a = sp1[0], S1b = sp1[1];
#pragma unroll
      for (int r = 0; r < 4; ++r) {
        acc0[r]     += s * P0[r]     + nzs * S0a[r];
        acc0[r + 4] += s * P0[r + 4] + nzs * S0b[r];
        acc1[r]     += s * P1[r]     + nzs * S1a[r];
        acc1[r + 4] += s * P1[r + 4] + nzs * S1b[r];
      }
    }
  }

  // ---- epilogue: C layout -> out, add bias
  const float bv = bias[n];
#pragma unroll
  for (int r = 0; r < 8; ++r) {
    const int m = half * 8 + r;
    out[m * OUT_F + n]        = acc0[r] + bv;
    out[(16 + m) * OUT_F + n] = acc1[r] + bv;
  }
}

extern "C" void kernel_launch(void* const* d_in, const int* in_sizes, int n_in,
                              void* d_out, int out_size, void* d_ws, size_t ws_size,
                              hipStream_t stream) {
  const float* inp     = (const float*)d_in[0];
  const int*   qweight = (const int*)d_in[1];
  const float* scales  = (const float*)d_in[2];
  const int*   qzeros  = (const int*)d_in[3];
  const float* bias    = (const float*)d_in[4];
  float* out = (float*)d_out;

  dim3 grid(OUT_F / NBLK);   // 86 blocks
  dim3 block(256);           // 8 waves (wave32)
  int4linear_wmma<<<grid, block, 0, stream>>>(inp, qweight, scales, qzeros, bias, out);
}
